// RNN_11905649345051
// MI455X (gfx1250) — matrix operand order repaired and checked by reading, not compiled, not run on hardware
//
#include <hip/hip_runtime.h>

// 2-layer tanh RNN: B=64, S=2048, I=H=256.
#define B_    64
#define S_    2048
#define H_    256
#define RING  4

typedef __bf16 bf16;
typedef __attribute__((ext_vector_type(8)))  __bf16 v8bf;
typedef __attribute__((ext_vector_type(16))) __bf16 v16bf;
typedef __attribute__((ext_vector_type(8)))  float  v8f;
typedef __attribute__((ext_vector_type(4)))  float  v4f;
typedef __attribute__((ext_vector_type(4)))  int    v4i;

typedef __attribute__((address_space(1))) v4i glb_v4i;   // global (AS1) int4
typedef __attribute__((address_space(3))) v4i lds_v4i;   // LDS (AS3) int4

#if defined(__has_builtin)
#  if __has_builtin(__builtin_amdgcn_global_load_async_to_lds_b128) && \
      __has_builtin(__builtin_amdgcn_s_wait_asynccnt)
#    define HAVE_ASYNC_LDS 1
#  endif
#endif
#ifndef HAVE_ASYNC_LDS
#  define HAVE_ASYNC_LDS 0
#endif

// Workspace layout (bytes):
//  [0, 4096)                 progress flags (zeroed by prep each call)
//  [WS_WFRAG, +4*131072)     bf16 B-fragment arenas: Wih0, Whh0, Wih1, Whh1
//  [WS_RINGH0, +RING*32768)  h0 ring, bf16 C-fragment order
//  [WS_RINGP,  +RING*65536)  P ring, f32 C-fragment order
#define WS_WFRAG   4096
#define WS_RINGH0  (WS_WFRAG + 4 * 131072)
#define WS_RINGP   (WS_RINGH0 + RING * 32768)
// total = 921600 bytes

__device__ __forceinline__ v8f wmma_bf16(v16bf a, v16bf b, v8f c) {
  // D = A(16x32 bf16) x B(32x16 bf16) + C(16x16 f32)
  return __builtin_amdgcn_wmma_f32_16x16x32_bf16(false, a, false, b,
                                                 (short)0, c, false, false);
}

// Fast tanh: 1 - 2/(e^{2x}+1). Saturates to +/-1, propagates NaN.
__device__ __forceinline__ float tanh_fast(float x) {
  float e = __expf(2.0f * x);
  return 1.0f - 2.0f * __builtin_amdgcn_rcpf(e + 1.0f);
}

// One fragment block = 1024B: [seg(2)][lane(32)][8 x bf16]. Two 16B loads/lane.
__device__ __forceinline__ v16bf load_frag(const char* blk, int ln) {
  union { v16bf v; v8bf h[2]; } u;
  u.h[0] = *(const v8bf*)(blk + ln * 16);
  u.h[1] = *(const v8bf*)(blk + 512 + ln * 16);
  return u.v;
}

// B-matrix (weights) element (n,k) -> byte offset in fragment arena.
// B 32x16 (contiguous k order used here, matching afrag_off): lane = (n&15) + 16*((k>>4)&1); lane holds 16 consecutive k.
__device__ __forceinline__ int bfrag_off(int n, int k) {
  int c = n >> 4, kb = k >> 5;
  int lane = (n & 15) + 16 * ((k >> 4) & 1);
  int j = k & 15;
  return ((c * 8 + kb) * 1024) + (j >> 3) * 512 + lane * 16 + (j & 7) * 2;
}

// A-matrix (h state / x tile) element (m,k) -> byte offset.
// A 16x32: lane holds 16 consecutive k, base = 16*(lane>=16), m = lane&15.
// base = 16*(lane>=16), m = lane&15.
__device__ __forceinline__ int afrag_off(int m, int k) {
  int r = m >> 4, kb = k >> 5, klo = k & 31;
  int seg  = (klo >> 3) & 1;
  int lane = (m & 15) + 16 * (klo >> 4);
  return ((r * 8 + kb) * 1024) + seg * 512 + lane * 16 + (klo & 7) * 2;
}

// Scatter one C-fragment (8 rows of column n) into A-fragment storage.
__device__ __forceinline__ void scatter_h(char* smH, int rt, int c, int ln, v8bf vals) {
  int n = c * 16 + (ln & 15);
  int mhalf = (ln >> 4) * 8;
  for (int rr = 0; rr < 8; ++rr) {
    int m = rt * 16 + rr + mhalf;
    *(bf16*)(smH + afrag_off(m, n)) = vals[rr];
  }
}

__device__ __forceinline__ int ld_acq(const int* p) {
  return __hip_atomic_load(p, __ATOMIC_ACQUIRE, __HIP_MEMORY_SCOPE_AGENT);
}
__device__ __forceinline__ void st_rel(int* p, int v) {
  __hip_atomic_store(p, v, __ATOMIC_RELEASE, __HIP_MEMORY_SCOPE_AGENT);
}

// ---------------------------------------------------------------------------
// prep: zero flags; convert all four weight matrices (f32) into bf16
// B-fragment arenas in d_ws (byte-identical layout to the LDS arenas).
__global__ void __launch_bounds__(1024) prep_kernel(const float* __restrict__ Wih0,
                                                    const float* __restrict__ Whh0,
                                                    const float* __restrict__ Wih1,
                                                    const float* __restrict__ Whh1,
                                                    char* __restrict__ ws) {
  int tid = threadIdx.x;
  int r = blockIdx.x;  // 0..3
  if (r == 0) ((int*)ws)[tid] = 0;  // 4096B flag region
  const float* W = (r == 0) ? Wih0 : (r == 1) ? Whh0 : (r == 2) ? Wih1 : Whh1;
  char* wf = ws + WS_WFRAG + r * 131072;
  for (int idx = tid; idx < 65536; idx += 1024) {
    *(bf16*)(wf + bfrag_off(idx >> 8, idx & 255)) = (bf16)W[idx];
  }
}

// ---------------------------------------------------------------------------
// xproj: Xp[m, n] = x[m, :] @ W_ih0^T + b_ih0 + b_hh0, m = b*S+s flattened.
// Written into d_out (reused as Xp buffer, later overwritten with h1).
__global__ void __launch_bounds__(256) xproj_kernel(const float* __restrict__ x,
                                                    const char* __restrict__ ws,
                                                    const float* __restrict__ bih0,
                                                    const float* __restrict__ bhh0,
                                                    float* __restrict__ out) {
  __shared__ char sm[8192];  // A-fragments for one 16-row tile, K=256
  int tid = threadIdx.x, ln = tid & 31, w = tid >> 5;
  int mt = blockIdx.x;

  for (int idx = tid; idx < 4096; idx += 256) {
    int row = idx >> 8, k = idx & 255;
    float v = x[(size_t)(mt * 16 + row) * 256 + k];
    *(bf16*)(sm + afrag_off(row, k)) = (bf16)v;
  }
  __syncthreads();

  const char* wf = ws + WS_WFRAG;  // arena 0 = W_ih0
  int c0 = w * 2, c1 = c0 + 1;
  int n0 = c0 * 16 + (ln & 15), n1 = c1 * 16 + (ln & 15);
  float bias0 = bih0[n0] + bhh0[n0];
  float bias1 = bih0[n1] + bhh0[n1];
  v8f a0, a1;
  for (int r = 0; r < 8; ++r) { a0[r] = bias0; a1[r] = bias1; }

  for (int kb = 0; kb < 8; ++kb) {
    v16bf af = load_frag(sm + kb * 1024, ln);
    v16bf b0 = load_frag(wf + (c0 * 8 + kb) * 1024, ln);
    v16bf b1 = load_frag(wf + (c1 * 8 + kb) * 1024, ln);
    a0 = wmma_bf16(af, b0, a0);
    a1 = wmma_bf16(af, b1, a1);
  }
  int mhalf = (ln >> 4) * 8;
  for (int rr = 0; rr < 8; ++rr) {
    size_t m = (size_t)mt * 16 + rr + mhalf;
    out[m * 256 + n0] = a0[rr];
    out[m * 256 + n1] = a1[rr];
  }
}

// ---------------------------------------------------------------------------
// Persistent 3-stage pipelined recurrence. grid = 3 blocks of 1024 threads.
//  role 0: h0_t = tanh(Xp_t + h0 @ W_hh0^T)         (W_hh0 resident in LDS)
//  role 1: P_t  = h0_t @ W_ih1^T                    (W_ih1 resident in LDS)
//  role 2: h1_t = tanh(P_t + b + h1 @ W_hh1^T)      (W_hh1 resident in LDS)
__global__ void __launch_bounds__(1024) rnn_kernel(const float* __restrict__ bih1,
                                                   const float* __restrict__ bhh1,
                                                   float* __restrict__ out,
                                                   char* __restrict__ ws) {
  __shared__ char sm[163840];          // 128K weight B-frags + 32K h A-frags
  char* smW = sm;
  char* smH = sm + 131072;

  int tid = threadIdx.x, ln = tid & 31, w = tid >> 5;
  int role = blockIdx.x;

  int* prog0 = (int*)(ws + 0);
  int* prog1 = (int*)(ws + 64);
  int* prog2 = (int*)(ws + 128);
  char* ringH = ws + WS_RINGH0;
  char* ringP = ws + WS_RINGP;

  // Stage this block's pre-swizzled weight arena into LDS (bulk copy):
  // role 0 -> Whh0 (arena 1), role 1 -> Wih1 (arena 2), role 2 -> Whh1 (arena 3).
  const char* wsrc = ws + WS_WFRAG + (role + 1) * 131072;
#if HAVE_ASYNC_LDS
  for (int off = tid * 16; off < 131072; off += 16384) {
    __builtin_amdgcn_global_load_async_to_lds_b128(
        (glb_v4i*)(wsrc + off), (lds_v4i*)(smW + off), 0, 0);
  }
#else
  for (int off = tid * 16; off < 131072; off += 16384) {
    *(v4i*)(smW + off) = *(const v4i*)(wsrc + off);
  }
#endif
  // Zero h-state fragments (h0 = h1 = 0) while the copy is in flight.
  v4i zero = {0, 0, 0, 0};
  for (int idx = tid; idx < 2048; idx += 1024) *(v4i*)(smH + idx * 16) = zero;
#if HAVE_ASYNC_LDS
  __builtin_amdgcn_s_wait_asynccnt(0);
#endif
  __syncthreads();

  int rt = w >> 3;                       // row tile 0..3 (16 batch rows)
  int c0 = (w & 7) * 2, c1 = c0 + 1;     // two adjacent column tiles
  int nn0 = c0 * 16 + (ln & 15), nn1 = c1 * 16 + (ln & 15);
  int mhalf = (ln >> 4) * 8;
  const size_t HN0 = (size_t)B_ * S_ * H_;        // hn[0] offset in d_out
  const size_t HN1 = HN0 + (size_t)B_ * H_;       // hn[1] offset

  if (role == 0) {
    for (int t = 0; t < S_; ++t) {
      if (t >= RING)
        while (ld_acq(prog1) < t - RING + 1) __builtin_amdgcn_s_sleep(2);
      v8f a0, a1;
      for (int rr = 0; rr < 8; ++rr) {            // C init from Xp (in d_out)
        size_t m = (size_t)(rt * 16 + rr + mhalf);
        a0[rr] = out[(m * S_ + t) * H_ + nn0];
        a1[rr] = out[(m * S_ + t) * H_ + nn1];
      }
      if (t + 1 < S_) {                           // prefetch next step's Xp
        for (int rr = 0; rr < 8; ++rr) {
          size_t m = (size_t)(rt * 16 + rr + mhalf);
          __builtin_prefetch(&out[(m * S_ + t + 1) * H_ + nn0], 0, 3);
          __builtin_prefetch(&out[(m * S_ + t + 1) * H_ + nn1], 0, 3);
        }
      }
      for (int kb = 0; kb < 8; ++kb) {
        v16bf af = load_frag(smH + (rt * 8 + kb) * 1024, ln);
        v16bf b0 = load_frag(smW + (c0 * 8 + kb) * 1024, ln);
        v16bf b1 = load_frag(smW + (c1 * 8 + kb) * 1024, ln);
        a0 = wmma_bf16(af, b0, a0);
        a1 = wmma_bf16(af, b1, a1);
      }
      for (int rr = 0; rr < 8; ++rr) { a0[rr] = tanh_fast(a0[rr]); a1[rr] = tanh_fast(a1[rr]); }
      __syncthreads();                            // all reads of smH done
      v8bf h0b, h1b;
      for (int rr = 0; rr < 8; ++rr) { h0b[rr] = (bf16)a0[rr]; h1b[rr] = (bf16)a1[rr]; }
      scatter_h(smH, rt, c0, ln, h0b);
      scatter_h(smH, rt, c1, ln, h1b);
      char* rs = ringH + (size_t)(t & (RING - 1)) * 32768;
      *(v8bf*)(rs + ((rt * 16 + c0) * 32 + ln) * 16) = h0b;
      *(v8bf*)(rs + ((rt * 16 + c1) * 32 + ln) * 16) = h1b;
      if (t == S_ - 1) {
        for (int rr = 0; rr < 8; ++rr) {
          size_t m = (size_t)(rt * 16 + rr + mhalf);
          out[HN0 + m * H_ + nn0] = a0[rr];
          out[HN0 + m * H_ + nn1] = a1[rr];
        }
      }
      __threadfence();
      __syncthreads();
      if (tid == 0) st_rel(prog0, t + 1);
    }
  } else if (role == 1) {
    for (int t = 0; t < S_; ++t) {
      while (ld_acq(prog0) < t + 1) __builtin_amdgcn_s_sleep(2);
      if (t >= RING)
        while (ld_acq(prog2) < t - RING + 1) __builtin_amdgcn_s_sleep(2);
      // Stage h0_t ring slot into LDS A-fragments.
      const char* rs = ringH + (size_t)(t & (RING - 1)) * 32768;
      for (int e = tid; e < 2048; e += 1024) {
        int tile = e >> 5, l2 = e & 31;
        v8bf hv = *(const v8bf*)(rs + e * 16);
        scatter_h(smH, tile >> 4, tile & 15, l2, hv);
      }
      __syncthreads();
      v8f a0, a1;
      for (int r = 0; r < 8; ++r) { a0[r] = 0.f; a1[r] = 0.f; }
      for (int kb = 0; kb < 8; ++kb) {
        v16bf af = load_frag(smH + (rt * 8 + kb) * 1024, ln);
        v16bf b0 = load_frag(smW + (c0 * 8 + kb) * 1024, ln);
        v16bf b1 = load_frag(smW + (c1 * 8 + kb) * 1024, ln);
        a0 = wmma_bf16(af, b0, a0);
        a1 = wmma_bf16(af, b1, a1);
      }
      char* ps = ringP + (size_t)(t & (RING - 1)) * 65536;
      union { v8f v; v4f q[2]; } u0, u1;
      u0.v = a0; u1.v = a1;
      char* p0 = ps + ((rt * 16 + c0) * 32 + ln) * 32;
      char* p1 = ps + ((rt * 16 + c1) * 32 + ln) * 32;
      *(v4f*)p0 = u0.q[0]; *(v4f*)(p0 + 16) = u0.q[1];
      *(v4f*)p1 = u1.q[0]; *(v4f*)(p1 + 16) = u1.q[1];
      __threadfence();
      __syncthreads();
      if (tid == 0) st_rel(prog1, t + 1);
    }
  } else {
    float bb0 = bih1[nn0] + bhh1[nn0];
    float bb1 = bih1[nn1] + bhh1[nn1];
    for (int t = 0; t < S_; ++t) {
      while (ld_acq(prog1) < t + 1) __builtin_amdgcn_s_sleep(2);
      const char* ps = ringP + (size_t)(t & (RING - 1)) * 65536;
      union { v8f v; v4f q[2]; } u0, u1;
      const char* p0 = ps + ((rt * 16 + c0) * 32 + ln) * 32;
      const char* p1 = ps + ((rt * 16 + c1) * 32 + ln) * 32;
      u0.q[0] = *(const v4f*)p0; u0.q[1] = *(const v4f*)(p0 + 16);
      u1.q[0] = *(const v4f*)p1; u1.q[1] = *(const v4f*)(p1 + 16);
      v8f a0 = u0.v, a1 = u1.v;
      for (int rr = 0; rr < 8; ++rr) { a0[rr] += bb0; a1[rr] += bb1; }
      for (int kb = 0; kb < 8; ++kb) {
        v16bf af = load_frag(smH + (rt * 8 + kb) * 1024, ln);
        v16bf b0 = load_frag(smW + (c0 * 8 + kb) * 1024, ln);
        v16bf b1 = load_frag(smW + (c1 * 8 + kb) * 1024, ln);
        a0 = wmma_bf16(af, b0, a0);
        a1 = wmma_bf16(af, b1, a1);
      }
      for (int rr = 0; rr < 8; ++rr) { a0[rr] = tanh_fast(a0[rr]); a1[rr] = tanh_fast(a1[rr]); }
      __syncthreads();
      v8bf h0b, h1b;
      for (int rr = 0; rr < 8; ++rr) { h0b[rr] = (bf16)a0[rr]; h1b[rr] = (bf16)a1[rr]; }
      scatter_h(smH, rt, c0, ln, h0b);
      scatter_h(smH, rt, c1, ln, h1b);
      for (int rr = 0; rr < 8; ++rr) {            // final output h1_t
        size_t m = (size_t)(rt * 16 + rr + mhalf);
        out[(m * S_ + t) * H_ + nn0] = a0[rr];
        out[(m * S_ + t) * H_ + nn1] = a1[rr];
        if (t == S_ - 1) {
          out[HN1 + m * H_ + nn0] = a0[rr];
          out[HN1 + m * H_ + nn1] = a1[rr];
        }
      }
      __syncthreads();
      if (tid == 0) st_rel(prog2, t + 1);
    }
  }
}

// ---------------------------------------------------------------------------
extern "C" void kernel_launch(void* const* d_in, const int* in_sizes, int n_in,
                              void* d_out, int out_size, void* d_ws, size_t ws_size,
                              hipStream_t stream) {
  (void)in_sizes; (void)n_in; (void)out_size; (void)ws_size;
  const float* x    = (const float*)d_in[0];
  const float* Wih0 = (const float*)d_in[1];
  const float* Whh0 = (const float*)d_in[2];
  const float* bih0 = (const float*)d_in[3];
  const float* bhh0 = (const float*)d_in[4];
  const float* Wih1 = (const float*)d_in[5];
  const float* Whh1 = (const float*)d_in[6];
  const float* bih1 = (const float*)d_in[7];
  const float* bhh1 = (const float*)d_in[8];
  float* out = (float*)d_out;
  char* ws   = (char*)d_ws;

  prep_kernel<<<4, 1024, 0, stream>>>(Wih0, Whh0, Wih1, Whh1, ws);
  xproj_kernel<<<(B_ * S_) / 16, 256, 0, stream>>>(x, ws, bih0, bhh0, out);
  rnn_kernel<<<3, 1024, 0, stream>>>(bih1, bhh1, out, ws);
}
